// Attention_61675730371202
// MI455X (gfx1250) — matrix operand order repaired and checked by reading, not compiled, not run on hardware
//
#include <hip/hip_runtime.h>
#include <hip/hip_bf16.h>
#include <math.h>

typedef __attribute__((ext_vector_type(16))) _Float16 v16h;
typedef __attribute__((ext_vector_type(8)))  _Float16 v8h;
typedef __attribute__((ext_vector_type(8)))  float    v8f;
typedef __attribute__((ext_vector_type(4)))  float    v4f;

#define NB   8
#define SQL  2048
#define SKVL 2048
#define DINL 512
#define DOUTL 512

#define WMMA_F16(a,b,c) __builtin_amdgcn_wmma_f32_16x16x32_f16(false,(a),false,(b),(short)0,(c),false,false)

// ---- fragment loaders (CDNA5 wave32 WMMA layouts) ------------------------
// A 16x32 f16: lane l (row M=l%16, hlf=l/16) holds K = [8*hlf..+8) and [16+8*hlf..+8)
__device__ __forceinline__ v16h load_a_f32(const float* __restrict__ A, int lda,
                                           int row, int ks, int hlf) {
  const float* p0 = A + (size_t)row * lda + ks + 8 * hlf;
  v4f f0 = *(const v4f*)(p0);
  v4f f1 = *(const v4f*)(p0 + 4);
  v4f f2 = *(const v4f*)(p0 + 16);
  v4f f3 = *(const v4f*)(p0 + 20);
  v16h a;
#pragma unroll
  for (int j = 0; j < 4; ++j) {
    a[j]      = (_Float16)f0[j];
    a[4 + j]  = (_Float16)f1[j];
    a[8 + j]  = (_Float16)f2[j];
    a[12 + j] = (_Float16)f3[j];
  }
  return a;
}

__device__ __forceinline__ v16h load_a_f16(const _Float16* A, int lda,
                                           int row, int ks, int hlf) {
  const _Float16* p0 = A + (size_t)row * lda + ks + 8 * hlf;
  v8h h0 = *(const v8h*)(p0);
  v8h h1 = *(const v8h*)(p0 + 16);
  v16h a;
#pragma unroll
  for (int j = 0; j < 8; ++j) { a[j] = h0[j]; a[8 + j] = h1[j]; }
  return a;
}

// B 32x16 f16: lane l (col N=l%16, khf=l/16) holds K = [8*khf..+8) and [16+8*khf..+8).
// Bt is B^T stored row-major [N][K].
__device__ __forceinline__ v16h load_b_f16(const _Float16* __restrict__ Bt, int ldb,
                                           int ncol, int ks, int khf) {
  const _Float16* p0 = Bt + (size_t)ncol * ldb + ks + 8 * khf;
  v8h h0 = *(const v8h*)(p0);
  v8h h1 = *(const v8h*)(p0 + 16);
  return __builtin_shufflevector(h0, h1, 0,1,2,3,4,5,6,7,8,9,10,11,12,13,14,15);
}

// ---- small conversion kernels --------------------------------------------
__global__ __launch_bounds__(256) void transpose_w_kernel(const float* __restrict__ W,
                                                          _Float16* __restrict__ WT) {
  int i = blockIdx.x * 256 + threadIdx.x;    // 512*512 elements
  int rI = i >> 9, cO = i & 511;             // W[rI][cO]  (DIN x DOUT)
  WT[(size_t)cO * DINL + rI] = (_Float16)W[i];
}

__global__ __launch_bounds__(256) void cvt_f16_kernel(const float* __restrict__ X,
                                                      _Float16* __restrict__ Y, int n) {
  int i = blockIdx.x * 256 + threadIdx.x;
  if (i < n) Y[i] = (_Float16)X[i];
}

// ---- projection GEMM: Out = X(f32) @ W, W^T given in f16 ------------------
// each wave: 16x64 output tile (A fragment reused across 4 B tiles), K=512
__global__ __launch_bounds__(256) void proj_kernel(const float* __restrict__ X,
                                                   const _Float16* __restrict__ WT,
                                                   _Float16* __restrict__ Out,
                                                   int rowsPerBatch, int transposed) {
  int lane = threadIdx.x & 31, wave = threadIdx.x >> 5;
  int hlf = lane >> 4, ln = lane & 15;
  int gt = blockIdx.x * 8 + wave;            // 8192 wave-tiles
  int mt = gt >> 3, ng = gt & 7;
  int arow = mt * 16 + ln;
  v8f acc[4] = {};
#pragma unroll 4
  for (int kk = 0; kk < 16; ++kk) {
    int ks = kk * 32;
    v16h a = load_a_f32(X, DINL, arow, ks, hlf);
#pragma unroll
    for (int t = 0; t < 4; ++t) {
      v16h b = load_b_f16(WT, DINL, ng * 64 + t * 16 + ln, ks, hlf);
      acc[t] = WMMA_F16(a, b, acc[t]);
    }
  }
#pragma unroll
  for (int t = 0; t < 4; ++t)
#pragma unroll
    for (int r = 0; r < 8; ++r) {
      int row = mt * 16 + r + 8 * hlf;
      int col = ng * 64 + t * 16 + ln;
      _Float16 hv = (_Float16)acc[t][r];
      if (!transposed) {
        Out[(size_t)row * DOUTL + col] = hv;
      } else {                                // store v^T: [B][DOUT][SKV]
        int bb = row / rowsPerBatch, rb = row % rowsPerBatch;
        Out[((size_t)bb * DOUTL + col) * rowsPerBatch + rb] = hv;
      }
    }
}

// ---- M^T = (Woff @ v_b)^T in f16, per batch -------------------------------
__global__ __launch_bounds__(256) void mmat_kernel(const _Float16* __restrict__ Woffh,
                                                   const _Float16* __restrict__ vT,
                                                   _Float16* __restrict__ MT) {
  int lane = threadIdx.x & 31, wave = threadIdx.x >> 5;
  int hlf = lane >> 4, ln = lane & 15;
  int gt = blockIdx.x * 8 + wave;            // 2048 wave-tiles
  int b = gt >> 8, rem = gt & 255;
  int mt = rem >> 3, ng = rem & 7;
  const _Float16* Bt = vT + (size_t)b * DOUTL * SKVL;
  int arow = mt * 16 + ln;                   // DIN row
  v8f acc[4] = {};
  for (int kk = 0; kk < 64; ++kk) {          // K = SKV = 2048
    int ks = kk * 32;
    v16h a = load_a_f16(Woffh, SKVL, arow, ks, hlf);
#pragma unroll
    for (int t = 0; t < 4; ++t) {
      v16h bf = load_b_f16(Bt, SKVL, ng * 64 + t * 16 + ln, ks, hlf);
      acc[t] = WMMA_F16(a, bf, acc[t]);
    }
  }
#pragma unroll
  for (int t = 0; t < 4; ++t)
#pragma unroll
    for (int r = 0; r < 8; ++r) {
      int row = mt * 16 + r + 8 * hlf;       // i (DIN)
      int col = ng * 64 + t * 16 + ln;       // o (DOUT)
      MT[((size_t)b * DOUTL + col) * DINL + row] = (_Float16)acc[t][r];
    }
}

// ---- bias_out = offset @ M  (f32 output, full overwrite of d_out) ---------
__global__ __launch_bounds__(256) void bias_kernel(const float* __restrict__ offset,
                                                   const _Float16* __restrict__ MT,
                                                   float* __restrict__ out) {
  int lane = threadIdx.x & 31, wave = threadIdx.x >> 5;
  int hlf = lane >> 4, ln = lane & 15;
  int gt = blockIdx.x * 8 + wave;            // 8192 wave-tiles
  int mt = gt >> 3, ng = gt & 7;
  int bb = (mt * 16) >> 11;                  // batch (SQ=2048 rows each)
  const _Float16* Bt = MT + (size_t)bb * DOUTL * DINL;
  int arow = mt * 16 + ln;
  v8f acc[4] = {};
#pragma unroll 4
  for (int kk = 0; kk < 16; ++kk) {
    int ks = kk * 32;
    v16h a = load_a_f32(offset, DINL, arow, ks, hlf);
#pragma unroll
    for (int t = 0; t < 4; ++t) {
      v16h bf = load_b_f16(Bt, DINL, ng * 64 + t * 16 + ln, ks, hlf);
      acc[t] = WMMA_F16(a, bf, acc[t]);
    }
  }
#pragma unroll
  for (int t = 0; t < 4; ++t)
#pragma unroll
    for (int r = 0; r < 8; ++r) {
      int row = mt * 16 + r + 8 * hlf;
      int col = ng * 64 + t * 16 + ln;
      out[(size_t)row * DOUTL + col] = acc[t][r];
    }
}

// ---- flash attention: out += softmax(q k^T * s) @ v -----------------------
// block: 32 q-rows of one batch, 8 waves = (2 q-halves) x (4 kv/d slices)
#define LDQ 528    // 512 + 16 halves pad
#define LDP 80     // 64 + 16 halves pad
__global__ __launch_bounds__(256) void flash_kernel(const _Float16* __restrict__ qg_,
                                                    const _Float16* __restrict__ kg_,
                                                    const _Float16* __restrict__ vg_,
                                                    float* __restrict__ out) {
  __shared__ __align__(16) _Float16 qtile[32 * LDQ];
  __shared__ __align__(16) _Float16 ptile[32 * LDP];
  __shared__ float maxbuf[2][16][4];
  __shared__ float sumbuf[2][16][4];

  int tid = threadIdx.x;
  int lane = tid & 31, wave = tid >> 5;
  int wq = wave >> 2, wn = wave & 3;
  int hlf = lane >> 4, ln = lane & 15;

  int b = blockIdx.x >> 6;                   // 64 blocks per batch
  int qblk = (blockIdx.x & 63) * 32;
  const _Float16* qg = qg_ + ((size_t)b * SQL + qblk) * DOUTL;
  const _Float16* kg = kg_ + (size_t)b * SKVL * DOUTL;
  const _Float16* vg = vg_ + (size_t)b * DOUTL * SKVL;   // v^T [DOUT][SKV]

  // stage 32x512 q tile into LDS (padded rows)
#pragma unroll
  for (int it = 0; it < 8; ++it) {
    int cid = tid + 256 * it;                // 2048 chunks of 8 halves
    int row = cid >> 6, c8 = (cid & 63) * 8;
    *(v8h*)&qtile[row * LDQ + c8] = *(const v8h*)&qg[(size_t)row * DOUTL + c8];
  }
  // prefetch first kv chunk toward this WGP while q staging settles
  __builtin_prefetch(kg + (size_t)tid * 128, 0, 1);                // 64 rows * 1KB
  __builtin_prefetch(vg + (size_t)(2 * tid) * SKVL, 0, 1);         // v^T row heads
  __builtin_prefetch(vg + (size_t)(2 * tid + 1) * SKVL, 0, 1);
  __syncthreads();

  float m_i[8], l_i[8];
  v8f oacc[8] = {};                          // 16 rows x 128 d-cols per wave
#pragma unroll
  for (int r = 0; r < 8; ++r) { m_i[r] = -1e30f; l_i[r] = 0.f; }

  const float scale = 0.04419417382415922f;  // 1/sqrt(512)

  for (int c = 0; c < 32; ++c) {             // kv chunks of 64
    int kvbase = c * 64;

    // prefetch next chunk's k rows (64KB) and v^T row segments while computing
    if (c < 31) {
      const _Float16* knext = kg + (size_t)(kvbase + 64) * DOUTL;
      __builtin_prefetch(knext + (size_t)tid * 128, 0, 1);         // 256B/line/thread
      const _Float16* vnext = vg + (size_t)(kvbase + 64);
      __builtin_prefetch(vnext + (size_t)(2 * tid) * SKVL, 0, 1);
      __builtin_prefetch(vnext + (size_t)(2 * tid + 1) * SKVL, 0, 1);
    }

    // --- scores: 16x16 tile per wave, K = DOUT = 512 ---
    v8f s = {};
    int arowl = wq * 16 + ln;
    int ncol = kvbase + wn * 16 + ln;
#pragma unroll 4
    for (int kk = 0; kk < 16; ++kk) {
      int ks = kk * 32;
      v16h a  = load_a_f16(qtile, LDQ, arowl, ks, hlf);
      v16h bf = load_b_f16(kg, DOUTL, ncol, ks, hlf);
      s = WMMA_F16(a, bf, s);
    }
    float sv[8], mx[8];
#pragma unroll
    for (int r = 0; r < 8; ++r) {
      sv[r] = s[r] * scale;
      float v = sv[r];
      v = fmaxf(v, __shfl_xor(v, 1));
      v = fmaxf(v, __shfl_xor(v, 2));
      v = fmaxf(v, __shfl_xor(v, 4));
      v = fmaxf(v, __shfl_xor(v, 8));
      mx[r] = v;                              // row max over this wave's 16 cols
    }
    if (ln == 0)
#pragma unroll
      for (int r = 0; r < 8; ++r) maxbuf[wq][r + 8 * hlf][wn] = mx[r];
    __syncthreads();

    // combine across the 4 kv-slice waves; online-softmax update
    float alpha[8], mnew[8], p[8], rs[8];
#pragma unroll
    for (int r = 0; r < 8; ++r) {
      int row = r + 8 * hlf;
      float mc = fmaxf(fmaxf(maxbuf[wq][row][0], maxbuf[wq][row][1]),
                       fmaxf(maxbuf[wq][row][2], maxbuf[wq][row][3]));
      float mn = fmaxf(m_i[r], mc);
      alpha[r] = __expf(m_i[r] - mn);
      mnew[r] = mn;
    }
#pragma unroll
    for (int r = 0; r < 8; ++r) {
      p[r] = __expf(sv[r] - mnew[r]);
      float v = p[r];
      v += __shfl_xor(v, 1);
      v += __shfl_xor(v, 2);
      v += __shfl_xor(v, 4);
      v += __shfl_xor(v, 8);
      rs[r] = v;
    }
    if (ln == 0)
#pragma unroll
      for (int r = 0; r < 8; ++r) sumbuf[wq][r + 8 * hlf][wn] = rs[r];
    // write P (f16) to LDS for A-operand reload, rescale accumulators
#pragma unroll
    for (int r = 0; r < 8; ++r) {
      ptile[(wq * 16 + r + 8 * hlf) * LDP + wn * 16 + ln] = (_Float16)p[r];
      m_i[r] = mnew[r];
    }
#pragma unroll
    for (int n = 0; n < 8; ++n)
#pragma unroll
      for (int r = 0; r < 8; ++r) oacc[n][r] *= alpha[r];
    __syncthreads();

#pragma unroll
    for (int r = 0; r < 8; ++r) {
      int row = r + 8 * hlf;
      float sc = sumbuf[wq][row][0] + sumbuf[wq][row][1] +
                 sumbuf[wq][row][2] + sumbuf[wq][row][3];
      l_i[r] = l_i[r] * alpha[r] + sc;
    }
    // --- P @ V: 16x128 per wave, K = 64 (2 steps of 32) ---
#pragma unroll
    for (int ks2 = 0; ks2 < 2; ++ks2) {
      v16h a = load_a_f16(ptile, LDP, wq * 16 + ln, ks2 * 32, hlf);
#pragma unroll
      for (int n = 0; n < 8; ++n) {
        v16h bf = load_b_f16(vg, SKVL, wn * 128 + n * 16 + ln,
                             kvbase + ks2 * 32, hlf);
        oacc[n] = WMMA_F16(a, bf, oacc[n]);
      }
    }
    __syncthreads();                         // protect ptile/maxbuf for next chunk
  }

  // epilogue: normalize, add pre-written bias term, store f32
  float inv[8];
#pragma unroll
  for (int r = 0; r < 8; ++r) inv[r] = 1.0f / l_i[r];
#pragma unroll
  for (int n = 0; n < 8; ++n)
#pragma unroll
    for (int r = 0; r < 8; ++r) {
      int row = qblk + wq * 16 + r + 8 * hlf;
      int col = wn * 128 + n * 16 + ln;
      size_t idx = ((size_t)b * SQL + row) * DOUTL + col;
      out[idx] = out[idx] + oacc[n][r] * inv[r];
    }
}

// ---- host launch ----------------------------------------------------------
extern "C" void kernel_launch(void* const* d_in, const int* in_sizes, int n_in,
                              void* d_out, int out_size, void* d_ws, size_t ws_size,
                              hipStream_t stream) {
  (void)in_sizes; (void)n_in; (void)out_size; (void)ws_size;
  const float* query  = (const float*)d_in[0];
  const float* key    = (const float*)d_in[1];
  const float* value  = (const float*)d_in[2];
  const float* offset = (const float*)d_in[3];
  const float* Wq     = (const float*)d_in[4];
  const float* Wk     = (const float*)d_in[5];
  const float* Wv     = (const float*)d_in[6];
  const float* Woff   = (const float*)d_in[7];
  float* out = (float*)d_out;

  _Float16* qh    = (_Float16*)d_ws;                       // [B*SQ][DOUT]
  _Float16* kh    = qh  + (size_t)NB * SQL * DOUTL;        // [B*SKV][DOUT]
  _Float16* vTh   = kh  + (size_t)NB * SKVL * DOUTL;       // [B][DOUT][SKV]
  _Float16* WqT   = vTh + (size_t)NB * DOUTL * SKVL;       // [DOUT][DIN]
  _Float16* WkT   = WqT + (size_t)DOUTL * DINL;
  _Float16* WvT   = WkT + (size_t)DOUTL * DINL;
  _Float16* Woffh = WvT + (size_t)DOUTL * DINL;            // [DIN][SKV]
  _Float16* MT    = Woffh + (size_t)DINL * SKVL;           // [B][DOUT][DIN]

  transpose_w_kernel<<<1024, 256, 0, stream>>>(Wq, WqT);
  transpose_w_kernel<<<1024, 256, 0, stream>>>(Wk, WkT);
  transpose_w_kernel<<<1024, 256, 0, stream>>>(Wv, WvT);
  cvt_f16_kernel<<<(DINL * SKVL + 255) / 256, 256, 0, stream>>>(Woff, Woffh, DINL * SKVL);

  proj_kernel<<<1024, 256, 0, stream>>>(query, WqT, qh,  SQL,  0);
  proj_kernel<<<1024, 256, 0, stream>>>(key,   WkT, kh,  SKVL, 0);
  proj_kernel<<<1024, 256, 0, stream>>>(value, WvT, vTh, SKVL, 1);   // store v^T

  mmat_kernel<<<256, 256, 0, stream>>>(Woffh, vTh, MT);
  bias_kernel<<<1024, 256, 0, stream>>>(offset, MT, out);            // full overwrite
  flash_kernel<<<512, 256, 0, stream>>>(qh, kh, vTh, out);           // += attention
}
